// CorrNodeEncoding1G_11596411699776
// MI455X (gfx1250) — matrix operand order repaired and checked by reading, not compiled, not run on hardware
//
#include <hip/hip_runtime.h>
#include <hip/hip_bf16.h>

typedef __attribute__((ext_vector_type(16))) _Float16     v16h;
typedef __attribute__((ext_vector_type(8)))  float        v8f;
typedef __attribute__((ext_vector_type(4)))  unsigned int v4u;

#define NROWS 100000
#define SCOLS 1024
#define DDIM  512
#define FEATK 32          // K padded 20 -> 32 for one 16x16x32 WMMA
#define LNEPS 1e-5f
#define LDS_STRIDE 516    // 516 % 64 == 4 ; 8*516 % 64 == 32 -> no bank conflict between lane halves

__device__ __forceinline__ float wave_sum(float v) {
#pragma unroll
  for (int o = 16; o >= 1; o >>= 1) v += __shfl_xor(v, o, 32);
  return v;
}
__device__ __forceinline__ float wave_max(float v) {
#pragma unroll
  for (int o = 16; o >= 1; o >>= 1) v = fmaxf(v, __shfl_xor(v, o, 32));
  return v;
}
__device__ __forceinline__ float wave_min(float v) {
#pragma unroll
  for (int o = 16; o >= 1; o >>= 1) v = fminf(v, __shfl_xor(v, o, 32));
  return v;
}

// ---------------------------------------------------------------------------
// Kernel 0: repack proj_w [20 x 512] f32 -> Bt[n][k] f16, K padded to 32.
// Bt[n*32 + k] = (k < 20) ? proj_w[k*512 + n] : 0
// ---------------------------------------------------------------------------
__global__ __launch_bounds__(256) void k_prepB(const float* __restrict__ W,
                                               _Float16* __restrict__ Bt) {
  int t = blockIdx.x * 256 + threadIdx.x;    // 0 .. 512*32-1
  int n = t >> 5;
  int k = t & 31;
  float v = (k < 20) ? W[k * DDIM + n] : 0.0f;
  Bt[t] = (_Float16)v;
}

// ---------------------------------------------------------------------------
// Kernel 1: per-row stats of C (one wave32 per row) + RBF features.
// zA[row][0..31] f16 : [mean, max, min, std, rbf(mean)x8, rbf(max)x8, 0 x12]
// ---------------------------------------------------------------------------
__global__ __launch_bounds__(256) void k_stats(const float* __restrict__ C,
                                               const float* __restrict__ centers,
                                               const float* __restrict__ widths,
                                               _Float16* __restrict__ zA) {
  const int lane = threadIdx.x & 31;
  const int wave = threadIdx.x >> 5;
  const int row  = blockIdx.x * 8 + wave;            // N/8 = 12500 blocks exactly

  const float4* c4 = (const float4*)(C + (size_t)row * SCOLS);
  float vals[32];
  float s = 0.0f, mx = -1e30f, mn = 1e30f;
#pragma unroll
  for (int i = 0; i < 8; ++i) {
    float4 v = c4[i * 32 + lane];                    // coalesced 512B / wave
    float a = fminf(fmaxf(v.x, 0.0f), 1.0f);
    float b = fminf(fmaxf(v.y, 0.0f), 1.0f);
    float c = fminf(fmaxf(v.z, 0.0f), 1.0f);
    float d = fminf(fmaxf(v.w, 0.0f), 1.0f);
    vals[4 * i + 0] = a; vals[4 * i + 1] = b;
    vals[4 * i + 2] = c; vals[4 * i + 3] = d;
    s  += (a + b) + (c + d);
    mx  = fmaxf(mx, fmaxf(fmaxf(a, b), fmaxf(c, d)));
    mn  = fminf(mn, fminf(fminf(a, b), fminf(c, d)));
  }
  s  = wave_sum(s);
  mx = wave_max(mx);
  mn = wave_min(mn);
  const float mean = s * (1.0f / SCOLS);

  float ss = 0.0f;                                   // exact two-pass variance
#pragma unroll
  for (int i = 0; i < 32; ++i) { float d = vals[i] - mean; ss += d * d; }
  ss = wave_sum(ss);
  const float sd = sqrtf(ss * (1.0f / SCOLS));

  // feature t computed by lane t
  float f;
  const int t = lane;
  if      (t == 0) f = mean;
  else if (t == 1) f = mx;
  else if (t == 2) f = mn;
  else if (t == 3) f = sd;
  else if (t < 12) {
    float c = centers[t - 4], w = widths[t - 4];
    float d = (mean - c) / (w + 1e-6f);
    f = __expf(-0.5f * d * d);
  } else if (t < 20) {
    float c = centers[t - 12], w = widths[t - 12];
    float d = (mx - c) / (w + 1e-6f);
    f = __expf(-0.5f * d * d);
  } else {
    f = 0.0f;                                        // K padding
  }
  zA[(size_t)row * FEATK + t] = (_Float16)f;         // coalesced 64B / row
}

// ---------------------------------------------------------------------------
// Kernel 2: pe = z @ W + b via v_wmma_f32_16x16x32_f16, then LayerNorm,
// gate, mask, residual add. One block = one 16-row stripe, 8 waves x 4 tiles.
// ---------------------------------------------------------------------------
__global__ __launch_bounds__(256) void k_gemm_ln(
    const _Float16* __restrict__ zA, const _Float16* __restrict__ Bt,
    const float* __restrict__ x,   const float* __restrict__ pb,
    const float* __restrict__ gamma, const float* __restrict__ beta,
    const float* __restrict__ gatep, const unsigned char* __restrict__ mask,
    float* __restrict__ out) {
  __shared__ float pe[16][LDS_STRIDE];

  const int lane   = threadIdx.x & 31;
  const int wave   = threadIdx.x >> 5;
  const int half16 = lane >> 4;                      // 0: lanes 0-15, 1: 16-31
  const int m      = lane & 15;
  const int stripe = blockIdx.x;                     // N/16 = 6250 exactly

  // A fragment: row M = lane%16 of zA. ISA 16-bit A layout:
  //   lanes 0-15 hold K={0..7,16..23}, lanes 16-31 hold K={8..15,24..31}.
  const v4u* za4 = (const v4u*)(zA + ((size_t)stripe * 16 + m) * FEATK);
  union { v16h h; v4u u[2]; } A;
  A.u[0] = za4[half16];           // halfs [0..7] or [8..15]
  A.u[1] = za4[2 + half16];       // halfs [16..23] or [24..31]

  // 4 column tiles per wave; one WMMA each (K=32 covers all features)
#pragma unroll
  for (int jt = 0; jt < 4; ++jt) {
    const int j = wave * 4 + jt;                     // tile column 0..31
    const int n = j * 16 + m;                        // global column 0..511
    // B fragment: column n; same K layout as A (lanes 0-15 K={0..7,16..23}, 16-31 K={8..15,24..31})
    const v4u* bt4 = (const v4u*)(Bt + (size_t)n * FEATK);
    union { v16h h; v4u u[2]; } B;
    B.u[0] = bt4[half16];         // halfs [0..7] or [8..15]
    B.u[1] = bt4[2 + half16];

    v8f acc = {};
    acc = __builtin_amdgcn_wmma_f32_16x16x32_f16(
        /*neg_a=*/false, A.h, /*neg_b=*/false, B.h,
        /*c_mod=*/(short)0, acc, /*reuse_a=*/false, /*reuse_b=*/false);

    const float pbn = pb[n];                         // same column for all 8 regs
#pragma unroll
    for (int r = 0; r < 8; ++r) {                    // D layout: row = r + 8*half16
      pe[r + half16 * 8][n] = acc[r] + pbn;
    }
  }
  __syncthreads();

  // LayerNorm + gate + mask + residual. Wave w handles rows 2w, 2w+1.
  const float gate = *gatep;
#pragma unroll
  for (int rr = 0; rr < 2; ++rr) {
    const int    mrow = wave * 2 + rr;
    const size_t g    = (size_t)stripe * 16 + mrow;
    float v[16];
    float s = 0.0f;
#pragma unroll
    for (int t = 0; t < 16; ++t) { v[t] = pe[mrow][t * 32 + lane]; s += v[t]; }
    s = wave_sum(s);
    const float mu = s * (1.0f / DDIM);
    float ss = 0.0f;
#pragma unroll
    for (int t = 0; t < 16; ++t) { float d = v[t] - mu; ss += d * d; }
    ss = wave_sum(ss);
    const float rstd = rsqrtf(ss * (1.0f / DDIM) + LNEPS);
    const float msk  = mask[g] ? 0.0f : 1.0f;        // all-false in harness

    const float* xr   = x   + g * DDIM;
    float*       orow = out + g * DDIM;
#pragma unroll
    for (int t = 0; t < 16; ++t) {
      const int c = t * 32 + lane;                   // coalesced
      const float pn = (v[t] - mu) * rstd * gamma[c] + beta[c];
      orow[c] = xr[c] + msk * gate * pn;
    }
  }
}

extern "C" void kernel_launch(void* const* d_in, const int* in_sizes, int n_in,
                              void* d_out, int out_size, void* d_ws, size_t ws_size,
                              hipStream_t stream) {
  const float*         x       = (const float*)d_in[0];
  const float*         C       = (const float*)d_in[1];
  const unsigned char* mask    = (const unsigned char*)d_in[2]; // jnp bool = 1 byte
  const float*         W       = (const float*)d_in[3];
  const float*         pb      = (const float*)d_in[4];
  const float*         gamma   = (const float*)d_in[5];
  const float*         beta    = (const float*)d_in[6];
  const float*         gatep   = (const float*)d_in[7];
  const float*         centers = (const float*)d_in[8];
  const float*         widths  = (const float*)d_in[9];
  float*               out     = (float*)d_out;

  // workspace: zA (N x 32 f16 = 6.4 MB), Bt (512 x 32 f16 = 32 KB, 16B aligned)
  _Float16* zA = (_Float16*)d_ws;
  _Float16* Bt = zA + (size_t)NROWS * FEATK;

  k_prepB<<<(DDIM * FEATK) / 256, 256, 0, stream>>>(W, Bt);
  k_stats<<<NROWS / 8, 256, 0, stream>>>(C, centers, widths, zA);
  k_gemm_ln<<<NROWS / 16, 256, 0, stream>>>(zA, Bt, x, pb, gamma, beta,
                                            gatep, mask, out);
}
